// GATE_79534204387621
// MI455X (gfx1250) — matrix operand order repaired and checked by reading, not compiled, not run on hardware
//
#include <hip/hip_runtime.h>

// ---------------------------------------------------------------------------
// CDNA5 (gfx1250) implementation of the GAM/TDU reference.
// Heavy math:  alpha(bf16) @ HW1/OW2 via v_wmma_f32_16x16x32_bf16, with the
// B operand staged into LDS through the gfx1250 async path
// (global_load_async_to_lds_b128 + s_wait_asynccnt), double-buffered.
// Block = 8 waves = 4 M-tiles x 2 N-groups; B frags preloaded per iteration
// so the WMMA chain issues back-to-back after one ds_load clause.
// ---------------------------------------------------------------------------

typedef __attribute__((ext_vector_type(16))) __bf16 v16bf;
typedef __attribute__((ext_vector_type(8)))  __bf16 v8bf;
typedef __attribute__((ext_vector_type(8)))  float  v8f;

#define CT     4096
#define D_     64
#define DH_    128
#define NEMB   100000
#define NDRUGS 150

__device__ __forceinline__ __bf16 f2bf(float f) {
  unsigned u = __builtin_bit_cast(unsigned, f);
  unsigned r = u + 0x7FFFu + ((u >> 16) & 1u);         // round-to-nearest-even
  unsigned short s = (unsigned short)(r >> 16);
  return __builtin_bit_cast(__bf16, s);
}

// --------------------------- init: stamp = -1 ------------------------------
__global__ void init_stamp(int* __restrict__ stamp, int n) {
  int i = blockIdx.x * 256 + threadIdx.x;
  if (i < n) stamp[i] = -1;
}

// ---------------- HW1 = (embed[code] @ W1), written in B-frag layout -------
__global__ __launch_bounds__(128)
void hw1_kernel(const float* __restrict__ embed, const int* __restrict__ code_t,
                const float* __restrict__ W1, __bf16* __restrict__ Bsw) {
  __shared__ float h[D_];
  const int r = blockIdx.x, tid = threadIdx.x;
  const int c = code_t[r];
  if (tid < D_) h[tid] = embed[(size_t)c * D_ + tid];
  __syncthreads();
  float s = 0.f;
#pragma unroll 8
  for (int k = 0; k < D_; ++k) s += h[k] * W1[k * DH_ + tid];
  // swizzle: k-index = r, n-index = tid, NT = 8 tiles of N
  const int kt = r >> 5, kin = r & 31, khalf = kin >> 4, e = kin & 15;
  const int nt = tid >> 4, nl = tid & 15;
  const size_t idx = ((((size_t)kt * 8 + nt) * 32) + (khalf * 16 + nl)) * 16 + e;
  Bsw[idx] = f2bf(s);
}

// ------------- fused row softmax of adj -> alpha (bf16, row-major) ---------
__global__ __launch_bounds__(256)
void softmax_rows(const float* __restrict__ adj, __bf16* __restrict__ alpha) {
  __shared__ float row[CT];
  __shared__ float red[256];
  const int r = blockIdx.x, tid = threadIdx.x;
  const float* a = adj + (size_t)r * CT;
  float m = -3.402823466e38f;
  for (int i = tid; i < CT; i += 256) { float v = a[i]; row[i] = v; m = fmaxf(m, v); }
  red[tid] = m; __syncthreads();
  for (int s = 128; s > 0; s >>= 1) { if (tid < s) red[tid] = fmaxf(red[tid], red[tid + s]); __syncthreads(); }
  m = red[0]; __syncthreads();
  float sum = 0.f;
  for (int i = tid; i < CT; i += 256) { float e = __expf(row[i] - m); row[i] = e; sum += e; }
  red[tid] = sum; __syncthreads();
  for (int s = 128; s > 0; s >>= 1) { if (tid < s) red[tid] += red[tid + s]; __syncthreads(); }
  const float inv = 1.0f / red[0];
  __bf16* out = alpha + (size_t)r * CT;
  for (int i = tid; i < CT; i += 256) out[i] = f2bf(row[i] * inv);
}

// ------------------------ WMMA GEMM:  relu(A@B + bias) ---------------------
// A: [4096 x K] bf16 row-major (streamed; L2-resident).
// Bsw: B in fragment-swizzled layout; one 32-K strip staged per block into
// LDS via async-to-LDS, double-buffered; frags read back via ds_load_b128.
// Block = 8 waves = (8/NG) M-tiles x NG N-groups of NTW 16-col tiles each.
template <int NTW, int NG>
__global__ __launch_bounds__(256)
void wmma_gemm(const __bf16* __restrict__ A, const __bf16* __restrict__ Bsw,
               const float* __restrict__ bias, float* __restrict__ Out, int K) {
  constexpr int NTT   = NTW * NG;                    // total N tiles
  constexpr int STRIP = NTT * 512;                   // __bf16 elems per K-strip
  constexpr int CHUNKS = (STRIP * 2) / (256 * 16);   // 16B chunks per thread
  __shared__ __attribute__((aligned(32))) __bf16 Bsm[2][STRIP];

  const int tid  = threadIdx.x;
  const int lane = tid & 31;
  const int wave = tid >> 5;
  const int mw = wave / NG;                          // M-tile within block
  const int ng = wave % NG;                          // N-group within block
  const int m0 = (blockIdx.x * (8 / NG) + mw) * 16;
  const int hf = lane >> 4;          // which K-half of the A fragment
  const int rr = lane & 15;          // row within the 16-row tile

  // async-stage one B K-strip into LDS buffer b (per-lane 16B transfers)
  auto stage = [&](int kt, int b) {
    const __bf16* gsrc = Bsw + (size_t)kt * STRIP;
#pragma unroll
    for (int c = 0; c < CHUNKS; ++c) {
      const int elem = (tid + c * 256) * 8;          // 8 bf16 = 16 bytes
      unsigned loff = (unsigned)(size_t)&Bsm[b][elem];
      unsigned long long gaddr = (unsigned long long)(gsrc + elem);
      asm volatile("global_load_async_to_lds_b128 %0, %1, off"
                   :: "v"(loff), "v"(gaddr) : "memory");
    }
  };

  v8f acc[NTW];
#pragma unroll
  for (int n = 0; n < NTW; ++n)
#pragma unroll
    for (int j = 0; j < 8; ++j) acc[n][j] = 0.f;

  const int Ktiles = K >> 5;
  const __bf16* arow = A + (size_t)(m0 + rr) * K;

  auto compute = [&](int kt, int cur) {
    const int k0 = kt << 5;
    // A fragment: two aligned 16B global loads (elements 0..7 and 8..15)
    v8bf alo = *(const v8bf*)(arow + k0 + hf * 16);
    v8bf ahi = *(const v8bf*)(arow + k0 + hf * 16 + 8);
    v16bf af;
#pragma unroll
    for (int i = 0; i < 8; ++i) { af[i] = alo[i]; af[i + 8] = ahi[i]; }
    // preload ALL B fragments (one ds_load clause), then WMMA back-to-back
    const __bf16* bbase = &Bsm[cur][(ng * NTW) * 512 + lane * 16];
    v16bf bfr[NTW];
#pragma unroll
    for (int n = 0; n < NTW; ++n) bfr[n] = *(const v16bf*)(bbase + n * 512);
#pragma unroll
    for (int n = 0; n < NTW; ++n)
      acc[n] = __builtin_amdgcn_wmma_f32_16x16x32_bf16(
          false, af, false, bfr[n], (short)0, acc[n], false, false);
  };

  stage(0, 0);
  int kt = 0;
  for (; kt < Ktiles - 1; ++kt) {                    // branch-free hot loop
    const int cur = kt & 1;
    stage(kt + 1, cur ^ 1);
    asm volatile("s_wait_asynccnt %0" :: "n"(CHUNKS) : "memory");
    __syncthreads();                                 // strip kt visible
    compute(kt, cur);
    __syncthreads();                                 // readers done pre-restage
  }
  asm volatile("s_wait_asynccnt 0" ::: "memory");    // peeled last iteration
  __syncthreads();
  compute(kt, kt & 1);

  const int N = NTT * 16;
#pragma unroll
  for (int n = 0; n < NTW; ++n) {
    const int col = (ng * NTW + n) * 16 + (lane & 15);
    const float b = bias[col];
#pragma unroll
    for (int j = 0; j < 8; ++j) {
      const int row = m0 + (lane >> 4) * 8 + j;
      Out[(size_t)row * N + col] = fmaxf(acc[n][j] + b, 0.f);
    }
  }
}

// --------------- OW2 = (o @ W2), written in B-frag layout (NT=4) -----------
__global__ __launch_bounds__(64)
void ow2_kernel(const float* __restrict__ o, const float* __restrict__ W2,
                __bf16* __restrict__ Bsw) {
  __shared__ float orow[DH_];
  const int r = blockIdx.x, tid = threadIdx.x;
  orow[tid]      = o[(size_t)r * DH_ + tid];
  orow[tid + 64] = o[(size_t)r * DH_ + tid + 64];
  __syncthreads();
  float s = 0.f;
#pragma unroll 8
  for (int k = 0; k < DH_; ++k) s += orow[k] * W2[k * D_ + tid];
  const int kt = r >> 5, kin = r & 31, khalf = kin >> 4, e = kin & 15;
  const int nt = tid >> 4, nl = tid & 15;
  const size_t idx = ((((size_t)kt * 4 + nt) * 32) + (khalf * 16 + nl)) * 16 + e;
  Bsw[idx] = f2bf(s);
}

// ------------------- TDU gated update + stamped scatter --------------------
__global__ __launch_bounds__(64)
void tdu_kernel(const float* __restrict__ g, const int* __restrict__ code_t,
                const float* __restrict__ RW, const float* __restrict__ Rb,
                const float* __restrict__ ZW, const float* __restrict__ Zb,
                const float* __restrict__ FtW, const float* __restrict__ Ftb,
                const float* __restrict__ GI, float* __restrict__ Fbuf,
                int* __restrict__ stamp, int t) {
  __shared__ float gr[D_], rg[D_];
  const int r = blockIdx.x, j = threadIdx.x;
  gr[j] = g[(size_t)r * D_ + j];
  __syncthreads();
  float sR = Rb[j], sZ = Zb[j];
#pragma unroll 8
  for (int k = 0; k < D_; ++k) { sR += gr[k] * RW[j * D_ + k]; sZ += gr[k] * ZW[j * D_ + k]; }
  const float R = 1.f / (1.f + __expf(-sR));
  const float Z = 1.f / (1.f + __expf(-sZ));
  rg[j] = R * gr[j] + gr[j];
  __syncthreads();
  float sF = Ftb[j];
#pragma unroll 8
  for (int k = 0; k < D_; ++k) sF += rg[k] * FtW[j * D_ + k];
  const float Ft  = tanhf(sF);
  const float F_t = (1.f - Z) * gr[j] + Z * Ft;
  const int c = code_t[r];
  const float gi = GI[(size_t)c * D_ + j];
  const int s = stamp[c];                         // read before any write
  const float prev = (s == t - 1) ? Fbuf[(size_t)c * D_ + j] : 0.f;
  Fbuf[(size_t)c * D_ + j] = gi * prev + (1.f - gi) * F_t;
  __syncthreads();
  if (j == 0) stamp[c] = t;
}

// ------------------------- Ot = [g_last | F_hat[code]] ---------------------
__global__ __launch_bounds__(128)
void otbuild_kernel(const float* __restrict__ g, const int* __restrict__ code_last,
                    const float* __restrict__ Fbuf, const int* __restrict__ stamp,
                    float* __restrict__ Ot) {
  const int r = blockIdx.x, j = threadIdx.x;
  float v;
  if (j < D_) {
    v = g[(size_t)r * D_ + j];
  } else {
    const int c = code_last[r];
    v = (stamp[c] == 3) ? Fbuf[(size_t)c * D_ + (j - D_)] : 0.f;
  }
  Ot[(size_t)r * (2 * D_) + j] = v;
}

// ------------- per-drug block: Y_hat[j] = sum_r gelu(Ot[r].fcW[j]+b) -------
__global__ __launch_bounds__(256)
void fc_kernel(const float* __restrict__ Ot, const float* __restrict__ fcW,
               const float* __restrict__ fcb, float* __restrict__ Yhat) {
  __shared__ float w[2 * D_];
  __shared__ float red[256];
  const int j = blockIdx.x, tid = threadIdx.x;
  if (tid < 2 * D_) w[tid] = fcW[(size_t)j * (2 * D_) + tid];
  __syncthreads();
  const float b = fcb[j];
  float acc = 0.f;
  for (int r = tid; r < CT; r += 256) {
    const float* row = Ot + (size_t)r * (2 * D_);
    float x = b;
#pragma unroll 8
    for (int k = 0; k < 2 * D_; ++k) x += row[k] * w[k];
    acc += 0.5f * x * (1.f + erff(x * 0.70710678118654752f));   // exact GELU
  }
  red[tid] = acc; __syncthreads();
  for (int s = 128; s > 0; s >>= 1) { if (tid < s) red[tid] += red[tid + s]; __syncthreads(); }
  if (tid == 0) Yhat[j] = red[0];
}

// --------------------------- sigmoid / threshold ---------------------------
__global__ void final_kernel(const float* __restrict__ Yhat, float* __restrict__ out) {
  const int j = blockIdx.x * blockDim.x + threadIdx.x;
  if (j < NDRUGS) {
    const float y  = Yhat[j];
    const float sg = 1.f / (1.f + __expf(-y));
    out[j]              = (sg > 0.2f) ? 1.f : 0.f;   // pred
    out[NDRUGS + j]     = sg;                        // sigged
    out[2 * NDRUGS + j] = y;                         // Y_hat
  }
}

// ---------------------------------------------------------------------------
extern "C" void kernel_launch(void* const* d_in, const int* in_sizes, int n_in,
                              void* d_out, int out_size, void* d_ws, size_t ws_size,
                              hipStream_t stream) {
  const float* adjs  = (const float*)d_in[0];
  const int*   codes = (const int*)  d_in[1];
  const float* embed = (const float*)d_in[2];
  const float* W1  = (const float*)d_in[3];
  const float* b1  = (const float*)d_in[4];
  const float* W2  = (const float*)d_in[5];
  const float* b2  = (const float*)d_in[6];
  const float* RW  = (const float*)d_in[7];
  const float* Rb  = (const float*)d_in[8];
  const float* ZW  = (const float*)d_in[9];
  const float* Zb  = (const float*)d_in[10];
  const float* FtW = (const float*)d_in[11];
  const float* Ftb = (const float*)d_in[12];
  const float* GI  = (const float*)d_in[13];
  const float* fcW = (const float*)d_in[14];
  const float* fcb = (const float*)d_in[15];
  float* out = (float*)d_out;

  char* ws = (char*)d_ws;
  size_t off = 0;
  auto alloc = [&](size_t bytes) -> char* {
    char* p = ws + off;
    off += (bytes + 255) & ~(size_t)255;
    return p;
  };
  __bf16* alpha = (__bf16*)alloc((size_t)CT * CT * 2);      // 32 MB, L2-resident
  __bf16* HW1   = (__bf16*)alloc((size_t)CT * DH_ * 2);
  float*  o     = (float*) alloc((size_t)CT * DH_ * 4);
  __bf16* OW2   = (__bf16*)alloc((size_t)CT * D_ * 2);
  float*  g     = (float*) alloc((size_t)CT * D_ * 4);
  float*  Fbuf  = (float*) alloc((size_t)NEMB * D_ * 4);
  int*    stamp = (int*)   alloc((size_t)NEMB * 4);
  float*  Ot    = (float*) alloc((size_t)CT * 2 * D_ * 4);
  float*  Yhat  = (float*) alloc(256 * 4);
  (void)ws_size; (void)in_sizes; (void)n_in; (void)out_size;

  init_stamp<<<(NEMB + 255) / 256, 256, 0, stream>>>(stamp, NEMB);

  for (int t = 0; t < 4; ++t) {
    const float* adj    = adjs  + (size_t)t * CT * CT;
    const int*   code_t = codes + (size_t)t * CT;

    hw1_kernel  <<<CT, 128, 0, stream>>>(embed, code_t, W1, HW1);
    softmax_rows<<<CT, 256, 0, stream>>>(adj, alpha);
    // o = relu(alpha @ HW1 + b1)      [4096 x 128]; 8 N-tiles, 2 groups
    wmma_gemm<4, 2><<<64, 256, 0, stream>>>(alpha, HW1, b1, o, CT);
    ow2_kernel  <<<CT, 64, 0, stream>>>(o, W2, OW2);
    // g = relu(alpha @ OW2 + b2)      [4096 x 64]; 4 N-tiles, 2 groups
    wmma_gemm<2, 2><<<64, 256, 0, stream>>>(alpha, OW2, b2, g, CT);
    tdu_kernel  <<<CT, 64, 0, stream>>>(g, code_t, RW, Rb, ZW, Zb, FtW, Ftb,
                                        GI, Fbuf, stamp, t);
  }

  otbuild_kernel<<<CT, 128, 0, stream>>>(g, codes + (size_t)3 * CT, Fbuf, stamp, Ot);
  fc_kernel     <<<NDRUGS, 256, 0, stream>>>(Ot, fcW, fcb, Yhat);
  final_kernel  <<<1, 256, 0, stream>>>(Yhat, out);
}
